// MyViT_30829275250887
// MI455X (gfx1250) — compile-verified, matrix operand order checked by reading, not run on hardware
//
#include <hip/hip_runtime.h>
#include <cmath>

typedef _Float16 half_t;
typedef __attribute__((ext_vector_type(16))) _Float16 v16h;
typedef __attribute__((ext_vector_type(8)))  float    v8f;
typedef __attribute__((ext_vector_type(4)))  int      i32x4;

#ifndef __has_builtin
#define __has_builtin(x) 0
#endif

// CDNA5 async global->LDS copy (ASYNCcnt-tracked). Probe round 2 confirmed the
// builtin exists and expects (int4 AS1*, ... AS3*, imm offset, imm cpol).
#if __has_builtin(__builtin_amdgcn_global_load_async_to_lds_b128)
#define HAVE_ASYNC_LDS 1
#else
#define HAVE_ASYNC_LDS 0
#endif

typedef __attribute__((address_space(1))) i32x4 as1_i32x4;
typedef __attribute__((address_space(3))) i32x4 as3_i32x4;

__device__ __forceinline__ void wait_async0() {
#if __has_builtin(__builtin_amdgcn_s_wait_asynccnt)
  __builtin_amdgcn_s_wait_asynccnt(0);
#else
  asm volatile("s_wait_asynccnt 0x0" ::: "memory");
#endif
}

#define TPB 256

// ---- model dims ----
#define NB    32
#define Pn    197
#define Ppad  224          // padded seq (multiple of 32)
#define Dm    768
#define Hh    12
#define Vd    64
#define MLPd  3072
#define NCn   1000
#define NTOK  (NB * Pn)    // 6304
#define NPATCH (NB * 196)  // 6272
#define BHn   (NB * Hh)    // 384
#define QKVC  (3 * Hh * Vd)

// ---- GEMM tiling: 256 threads = 8 waves (4 along M x 2 along N);
// block tile 128x128, wave tile 32x64, K-step 32, double-buffered LDS ----
#define BMT 128
#define BNT 128
#define BKT 32
#define LDSS 40   // BKT + 8 halves pad; row stride stays 16B aligned

// 16x32 f16 WMMA operand fragment from LDS (CDNA5 ISA 7.12.2 16-bit layout):
//   lane<16 : row=lane,    K = {0..7, 16..23}
//   lane>=16: row=lane-16, K = {8..15, 24..31}
__device__ __forceinline__ v16h ldfrag(const half_t* p) {
  union { v16h v; uint4 q[2]; } u;
  u.q[0] = *reinterpret_cast<const uint4*>(p);
  u.q[1] = *reinterpret_cast<const uint4*>(p + 16);
  return u.v;
}

// C = A(MxK f16 row-major) * Bt^T, Bt stored [N][K] f16. Batched over blockIdx.z;
// output base = (z/zdiv)*sC1 + (z%zdiv)*sC2.
// EPI: 0 store f32 | 1 f32 += | 2 f32 += acc+bias | 3 f16 gelu(acc+bias) | 4 store f16
template<int EPI>
__global__ __launch_bounds__(TPB) void gemm_wmma(
    const half_t* __restrict__ A,  int lda, long long sA,
    const half_t* __restrict__ Bt, int ldb, long long sB,
    void* __restrict__ Cv, int ldc, long long sC1, long long sC2, int zdiv,
    const float* __restrict__ bias, int M, int N, int K, int m_store)
{
  __shared__ half_t As[2][BMT * LDSS];
  __shared__ half_t Bs[2][BNT * LDSS];

  const int tid = threadIdx.x;
  const int z = blockIdx.z;
  A  += (long long)z * sA;
  Bt += (long long)z * sB;
  const long long cbase = (long long)(z / zdiv) * sC1 + (long long)(z % zdiv) * sC2;

  const int m0 = blockIdx.y * BMT;
  const int n0 = blockIdx.x * BNT;

  const int wave = tid >> 5;
  const int lane = tid & 31;
  const int wm = wave & 3;   // 4 waves along M (32 rows each)
  const int wn = wave >> 2;  // 2 waves along N (64 cols each)

  // global->LDS copy coords: 8 contiguous halves (b128) per thread per chunk
  const int arow = tid >> 2;          // 0..63
  const int acol = (tid << 3) & 31;   // 0,8,16,24

  const v8f vz = {0.f,0.f,0.f,0.f,0.f,0.f,0.f,0.f};
  v8f acc[2][4];
  #pragma unroll
  for (int mi = 0; mi < 2; ++mi)
    #pragma unroll
    for (int ni = 0; ni < 4; ++ni) acc[mi][ni] = vz;

#if HAVE_ASYNC_LDS
  auto issue = [&](int buf, int k0) {
    #pragma unroll
    for (int h = 0; h < 2; ++h) {
      const int r = arow + h * 64;
      if (m0 + r < M)
        __builtin_amdgcn_global_load_async_to_lds_b128(
            (as1_i32x4*)(A + (long long)(m0 + r) * lda + k0 + acol),
            (as3_i32x4*)(&As[buf][r * LDSS + acol]), 0, 0);
      if (n0 + r < N)
        __builtin_amdgcn_global_load_async_to_lds_b128(
            (as1_i32x4*)(Bt + (long long)(n0 + r) * ldb + k0 + acol),
            (as3_i32x4*)(&Bs[buf][r * LDSS + acol]), 0, 0);
    }
  };
#endif

  const int nk = K / BKT;
#if HAVE_ASYNC_LDS
  issue(0, 0);
#else
  const uint4 zero4 = make_uint4(0u, 0u, 0u, 0u);
#endif

  for (int t = 0; t < nk; ++t) {
#if HAVE_ASYNC_LDS
    wait_async0();          // own async copies for buffer t&1 have landed
    __syncthreads();        // every wave's copies have landed; prev readers done
    if (t + 1 < nk) issue((t + 1) & 1, (t + 1) * BKT);
    const half_t* ab = As[t & 1];
    const half_t* bb = Bs[t & 1];
#else
    {
      const int k0 = t * BKT;
      #pragma unroll
      for (int h = 0; h < 2; ++h) {
        const int r = arow + h * 64;
        uint4 va = zero4, vb = zero4;
        if (m0 + r < M) va = *reinterpret_cast<const uint4*>(A + (long long)(m0 + r) * lda + k0 + acol);
        if (n0 + r < N) vb = *reinterpret_cast<const uint4*>(Bt + (long long)(n0 + r) * ldb + k0 + acol);
        *reinterpret_cast<uint4*>(&As[0][r * LDSS + acol]) = va;
        *reinterpret_cast<uint4*>(&Bs[0][r * LDSS + acol]) = vb;
      }
      __syncthreads();
    }
    const half_t* ab = As[0];
    const half_t* bb = Bs[0];
#endif

    const int fo = (lane & 16) ? 8 : 0;
    const int fr = lane & 15;
    v16h af[2], bf[4];
    #pragma unroll
    for (int mi = 0; mi < 2; ++mi)
      af[mi] = ldfrag(ab + (wm * 32 + mi * 16 + fr) * LDSS + fo);
    #pragma unroll
    for (int ni = 0; ni < 4; ++ni)
      bf[ni] = ldfrag(bb + (wn * 64 + ni * 16 + fr) * LDSS + fo);

    #pragma unroll
    for (int mi = 0; mi < 2; ++mi)
      #pragma unroll
      for (int ni = 0; ni < 4; ++ni)
        acc[mi][ni] = __builtin_amdgcn_wmma_f32_16x16x32_f16(
            false, af[mi], false, bf[ni], (short)0, acc[mi][ni], false, false);

#if !HAVE_ASYNC_LDS
    __syncthreads();
#endif
  }

  // Epilogue. C/D 16x16 f32: VGPR r -> (M=r, N=lane) lanes0-15 / (M=r+8, N=lane-16)
  float*  Cf = reinterpret_cast<float*>(Cv)  + cbase;
  half_t* Ch = reinterpret_cast<half_t*>(Cv) + cbase;
  const int cn = lane & 15;
  const int ra = (lane & 16) ? 8 : 0;
  #pragma unroll
  for (int mi = 0; mi < 2; ++mi) {
    #pragma unroll
    for (int ni = 0; ni < 4; ++ni) {
      int gn = n0 + wn * 64 + ni * 16 + cn;
      if (gn >= N) continue;
      int gmb = m0 + wm * 32 + mi * 16 + ra;
      #pragma unroll
      for (int r = 0; r < 8; ++r) {
        int gm = gmb + r;
        if (gm >= m_store) continue;
        float v = acc[mi][ni][r];
        long long idx = (long long)gm * ldc + gn;
        if (EPI == 0) {
          Cf[idx] = v;
        } else if (EPI == 1) {
          Cf[idx] += v;
        } else if (EPI == 2) {
          Cf[idx] += v + bias[gn];
        } else if (EPI == 3) {
          float tv = v + bias[gn];
          Ch[idx] = (half_t)(0.5f * tv * (1.0f + erff(tv * 0.70710678118654752f)));
        } else {
          Ch[idx] = (half_t)v;
        }
      }
    }
  }
}

// fp32 [K][N] -> f16 [N][K], 32x32 LDS-tiled (coalesced both sides)
__global__ __launch_bounds__(TPB) void k_transpose_h(const float* __restrict__ in,
                                                     half_t* __restrict__ out,
                                                     int K, int N) {
  __shared__ half_t tile[32][33];
  const int k0 = blockIdx.x * 32, n0 = blockIdx.y * 32;
  const int tx = threadIdx.x & 31, ty = threadIdx.x >> 5;  // 32 x 8
  #pragma unroll
  for (int i = 0; i < 32; i += 8) {
    int k = k0 + ty + i, n = n0 + tx;
    tile[ty + i][tx] = (k < K && n < N) ? (half_t)in[(long long)k * N + n] : (half_t)0.f;
  }
  __syncthreads();
  #pragma unroll
  for (int i = 0; i < 32; i += 8) {
    int n = n0 + ty + i, k = k0 + tx;
    if (n < N && k < K) out[(long long)n * K + k] = tile[tx][ty + i];
  }
}

// images [32][3][224][224] -> f16 patch matrix [6272][768], col = c*256 + y*16 + x
__global__ void k_patchify(const float* __restrict__ img, half_t* __restrict__ out) {
  long long i = (long long)blockIdx.x * TPB + threadIdx.x;
  if (i >= (long long)NPATCH * Dm) return;
  int col = (int)(i % Dm);
  int row = (int)(i / Dm);
  int b = row / 196, pp = row % 196;
  int pi = pp / 14, pj = pp % 14;
  int c = col >> 8;
  int rem = col & 255;
  int y = rem >> 4, xc = rem & 15;
  out[i] = (half_t)img[(((long long)(b * 3 + c) * 224) + pi * 16 + y) * 224 + pj * 16 + xc];
}

// x[b,p,d] = (p==0 ? cls[d] : tok[b*196+p-1, d]) + pos[p, d]
__global__ void k_build_x(const float* __restrict__ tok, const float* __restrict__ cls,
                          const float* __restrict__ pos, float* __restrict__ x) {
  long long i = (long long)blockIdx.x * TPB + threadIdx.x;
  if (i >= (long long)NTOK * Dm) return;
  int d = (int)(i % Dm);
  int p = (int)((i / Dm) % Pn);
  int b = (int)(i / ((long long)Dm * Pn));
  float v = (p == 0) ? cls[d] : tok[(long long)(b * 196 + p - 1) * Dm + d];
  x[i] = v + pos[(long long)p * Dm + d];
}

// per-row LayerNorm (768), f16 output
__global__ __launch_bounds__(TPB) void k_layernorm_h(const float* __restrict__ x,
    const float* __restrict__ w, const float* __restrict__ bb, half_t* __restrict__ out) {
  __shared__ float r1[TPB];
  __shared__ float r2[TPB];
  int row = blockIdx.x, t = threadIdx.x;
  const float* xr = x + (long long)row * Dm;
  float a0 = xr[t], a1 = xr[t + 256], a2 = xr[t + 512];
  r1[t] = a0 + a1 + a2;
  r2[t] = a0 * a0 + a1 * a1 + a2 * a2;
  __syncthreads();
  for (int s = 128; s > 0; s >>= 1) {
    if (t < s) { r1[t] += r1[t + s]; r2[t] += r2[t + s]; }
    __syncthreads();
  }
  float mean = r1[0] * (1.0f / 768.0f);
  float var  = r2[0] * (1.0f / 768.0f) - mean * mean;
  float inv  = rsqrtf(var + 1e-5f);
  half_t* o = out + (long long)row * Dm;
  o[t]       = (half_t)((a0 - mean) * inv * w[t]       + bb[t]);
  o[t + 256] = (half_t)((a1 - mean) * inv * w[t + 256] + bb[t + 256]);
  o[t + 512] = (half_t)((a2 - mean) * inv * w[t + 512] + bb[t + 512]);
}

// qkv f32 [6304][2304] (col = h*192 + v*3 + {q,k,v}) -> q,k f16 [bh][Ppad][64] (q/8),
// v transposed f16 [bh][64][Ppad]. Padded rows zeroed.
__global__ void k_scatter_qkv(const float* __restrict__ qkv, half_t* __restrict__ q,
                              half_t* __restrict__ k, half_t* __restrict__ vt) {
  long long i = (long long)blockIdx.x * TPB + threadIdx.x;
  if (i >= (long long)BHn * Ppad * Vd) return;
  int v  = (int)(i & 63);
  int p  = (int)((i >> 6) % Ppad);
  int bh = (int)(i / ((long long)Ppad * Vd));
  int b = bh / Hh, h = bh % Hh;
  float qv = 0.f, kv = 0.f, vv = 0.f;
  if (p < Pn) {
    long long base = (long long)(b * Pn + p) * QKVC + h * (Vd * 3) + v * 3;
    qv = qkv[base] * 0.125f;   // V^-0.5 folded into q
    kv = qkv[base + 1];
    vv = qkv[base + 2];
  }
  long long qi = ((long long)bh * Ppad + p) * Vd + v;
  q[qi] = (half_t)qv;
  k[qi] = (half_t)kv;
  vt[((long long)bh * Vd + v) * Ppad + p] = (half_t)vv;
}

// row softmax over j<197 of scores f32 [bh][Ppad][Ppad] -> f16 probs (padded cols = 0)
__global__ __launch_bounds__(TPB) void k_attn_softmax(const float* __restrict__ S,
                                                      half_t* __restrict__ P) {
  __shared__ float red[TPB];
  int i = blockIdx.x, bh = blockIdx.y, j = threadIdx.x;
  const float* row = S + ((long long)bh * Ppad + i) * Ppad;
  float sv = (j < Pn) ? row[j] : -1e30f;
  red[j] = sv;
  __syncthreads();
  for (int s = 128; s > 0; s >>= 1) {
    if (j < s) red[j] = fmaxf(red[j], red[j + s]);
    __syncthreads();
  }
  float mx = red[0];
  __syncthreads();
  float e = (j < Pn) ? __expf(sv - mx) : 0.f;
  red[j] = e;
  __syncthreads();
  for (int s = 128; s > 0; s >>= 1) {
    if (j < s) red[j] += red[j + s];
    __syncthreads();
  }
  float inv = 1.0f / red[0];
  if (j < Ppad) P[((long long)bh * Ppad + i) * Ppad + j] = (half_t)(e * inv);
}

// classifier head + softmax, one block per batch row (tiny: 32x768x1000)
__global__ __launch_bounds__(TPB) void k_head_softmax(const float* __restrict__ x,
    const float* __restrict__ hw, const float* __restrict__ hb, float* __restrict__ out) {
  __shared__ float xr[Dm];
  __shared__ float lg[NCn];
  __shared__ float red[TPB];
  int b = blockIdx.x, t = threadIdx.x;
  const float* xc = x + (long long)b * Pn * Dm;  // cls token row
  xr[t] = xc[t]; xr[t + 256] = xc[t + 256]; xr[t + 512] = xc[t + 512];
  __syncthreads();
  for (int c = t; c < NCn; c += TPB) {
    float acc = hb[c];
    for (int d = 0; d < Dm; ++d) acc += xr[d] * hw[(long long)d * NCn + c];
    lg[c] = acc;
  }
  __syncthreads();
  float m = -1e30f;
  for (int c = t; c < NCn; c += TPB) m = fmaxf(m, lg[c]);
  red[t] = m;
  __syncthreads();
  for (int s = 128; s > 0; s >>= 1) { if (t < s) red[t] = fmaxf(red[t], red[t + s]); __syncthreads(); }
  float mx = red[0];
  __syncthreads();
  float sum = 0.f;
  for (int c = t; c < NCn; c += TPB) { float e = __expf(lg[c] - mx); lg[c] = e; sum += e; }
  red[t] = sum;
  __syncthreads();
  for (int s = 128; s > 0; s >>= 1) { if (t < s) red[t] += red[t + s]; __syncthreads(); }
  float inv = 1.0f / red[0];
  for (int c = t; c < NCn; c += TPB) out[(long long)b * NCn + c] = lg[c] * inv;
}

extern "C" void kernel_launch(void* const* d_in, const int* in_sizes, int n_in,
                              void* d_out, int out_size, void* d_ws, size_t ws_size,
                              hipStream_t stream) {
  const float* images = (const float*)d_in[0];
  const float* W_map  = (const float*)d_in[1];
  const float* cls    = (const float*)d_in[2];
  const float* pos    = (const float*)d_in[3];
  const float* ln1w   = (const float*)d_in[4];
  const float* ln1b   = (const float*)d_in[5];
  const float* qkvw   = (const float*)d_in[6];
  const float* mergew = (const float*)d_in[7];
  const float* ln2w   = (const float*)d_in[8];
  const float* ln2b   = (const float*)d_in[9];
  const float* w1     = (const float*)d_in[10];
  const float* b1     = (const float*)d_in[11];
  const float* w2     = (const float*)d_in[12];
  const float* b2     = (const float*)d_in[13];
  const float* hw     = (const float*)d_in[14];
  const float* hb     = (const float*)d_in[15];
  float* out = (float*)d_out;
  (void)in_sizes; (void)n_in; (void)out_size; (void)ws_size;

  size_t off = 0;
  auto take = [&](size_t bytes) -> char* {
    char* p = (char*)d_ws + off;
    off += (bytes + 255) & ~(size_t)255;
    return p;
  };
  float*  x_f   = (float*) take((size_t)NTOK * Dm * 4);
  half_t* hbuf  = (half_t*)take((size_t)NTOK * Dm * 2);
  float*  tok   = (float*) take((size_t)NPATCH * Dm * 4);
  half_t* wmapT = (half_t*)take((size_t)Dm * Dm * 2);
  half_t* qkvT  = (half_t*)take((size_t)Dm * QKVC * 2);
  half_t* mrgT  = (half_t*)take((size_t)Dm * Dm * 2);
  half_t* w1T   = (half_t*)take((size_t)Dm * MLPd * 2);
  half_t* w2T   = (half_t*)take((size_t)Dm * MLPd * 2);
  float*  qkvF  = (float*) take((size_t)NTOK * QKVC * 4);
  half_t* qH    = (half_t*)take((size_t)BHn * Ppad * Vd * 2);
  half_t* kH    = (half_t*)take((size_t)BHn * Ppad * Vd * 2);
  half_t* vT    = (half_t*)take((size_t)BHn * Vd * Ppad * 2);
  float*  scF   = (float*) take((size_t)BHn * Ppad * Ppad * 4);
  half_t* pH    = (half_t*)take((size_t)BHn * Ppad * Ppad * 2);
  half_t* oH    = (half_t*)take((size_t)NTOK * Dm * 2);
  half_t* gH    = (half_t*)take((size_t)NTOK * MLPd * 2);

  auto cdiv = [](long long a, long long b) { return (int)((a + b - 1) / b); };

  // ---- patch embedding ----
  k_transpose_h<<<dim3(Dm / 32, Dm / 32, 1), TPB, 0, stream>>>(W_map, wmapT, Dm, Dm);
  k_patchify<<<cdiv((long long)NPATCH * Dm, TPB), TPB, 0, stream>>>(images, hbuf);
  gemm_wmma<0><<<dim3(Dm / BNT, cdiv(NPATCH, BMT), 1), TPB, 0, stream>>>(
      hbuf, Dm, 0, wmapT, Dm, 0, tok, Dm, 0, 0, 1, nullptr, NPATCH, Dm, Dm, NPATCH);
  k_build_x<<<cdiv((long long)NTOK * Dm, TPB), TPB, 0, stream>>>(tok, cls, pos, x_f);

  // ---- 12 transformer blocks ----
  for (int l = 0; l < 12; ++l) {
    k_transpose_h<<<dim3(Dm / 32, QKVC / 32, 1), TPB, 0, stream>>>(
        qkvw + (long long)l * Dm * QKVC, qkvT, Dm, QKVC);
    k_transpose_h<<<dim3(Dm / 32, Dm / 32, 1), TPB, 0, stream>>>(
        mergew + (long long)l * Dm * Dm, mrgT, Dm, Dm);
    k_transpose_h<<<dim3(Dm / 32, MLPd / 32, 1), TPB, 0, stream>>>(
        w1 + (long long)l * Dm * MLPd, w1T, Dm, MLPd);
    k_transpose_h<<<dim3(MLPd / 32, Dm / 32, 1), TPB, 0, stream>>>(
        w2 + (long long)l * Dm * MLPd, w2T, MLPd, Dm);

    // LN1 -> qkv GEMM -> split/scale
    k_layernorm_h<<<NTOK, TPB, 0, stream>>>(x_f, ln1w + l * Dm, ln1b + l * Dm, hbuf);
    gemm_wmma<0><<<dim3(QKVC / BNT, cdiv(NTOK, BMT), 1), TPB, 0, stream>>>(
        hbuf, Dm, 0, qkvT, Dm, 0, qkvF, QKVC, 0, 0, 1, nullptr, NTOK, QKVC, Dm, NTOK);
    k_scatter_qkv<<<cdiv((long long)BHn * Ppad * Vd, TPB), TPB, 0, stream>>>(qkvF, qH, kH, vT);

    // scores = (q/8) k^T  (batched over 384 b*h)
    gemm_wmma<0><<<dim3(cdiv(Ppad, BNT), cdiv(Ppad, BMT), BHn), TPB, 0, stream>>>(
        qH, Vd, (long long)Ppad * Vd, kH, Vd, (long long)Ppad * Vd,
        scF, Ppad, (long long)Ppad * Ppad, 0, 1, nullptr, Ppad, Ppad, Vd, Ppad);
    k_attn_softmax<<<dim3(Ppad, BHn, 1), TPB, 0, stream>>>(scF, pH);

    // o = P v, scattered straight into [b, p, h*64+v] f16
    gemm_wmma<4><<<dim3(cdiv(Vd, BNT), cdiv(Ppad, BMT), BHn), TPB, 0, stream>>>(
        pH, Ppad, (long long)Ppad * Ppad, vT, Ppad, (long long)Vd * Ppad,
        oH, Dm, (long long)Pn * Dm, (long long)Vd, Hh, nullptr, Ppad, Vd, Ppad, Pn);

    // x += o @ merge
    gemm_wmma<1><<<dim3(Dm / BNT, cdiv(NTOK, BMT), 1), TPB, 0, stream>>>(
        oH, Dm, 0, mrgT, Dm, 0, x_f, Dm, 0, 0, 1, nullptr, NTOK, Dm, Dm, NTOK);

    // MLP: LN2 -> gelu(h2 w1 + b1) -> x += g w2 + b2
    k_layernorm_h<<<NTOK, TPB, 0, stream>>>(x_f, ln2w + l * Dm, ln2b + l * Dm, hbuf);
    gemm_wmma<3><<<dim3(MLPd / BNT, cdiv(NTOK, BMT), 1), TPB, 0, stream>>>(
        hbuf, Dm, 0, w1T, Dm, 0, gH, MLPd, 0, 0, 1, b1 + (long long)l * MLPd,
        NTOK, MLPd, Dm, NTOK);
    gemm_wmma<2><<<dim3(Dm / BNT, cdiv(NTOK, BMT), 1), TPB, 0, stream>>>(
        gH, MLPd, 0, w2T, MLPd, 0, x_f, Dm, 0, 0, 1, b2 + (long long)l * Dm,
        NTOK, Dm, MLPd, NTOK);
  }

  // ---- classifier head + softmax ----
  k_head_softmax<<<NB, TPB, 0, stream>>>(x_f, hw, hb, out);
}
